// DifferentiableFeatureAligner2_39152921870906
// MI455X (gfx1250) — matrix operand order repaired and checked by reading, not compiled, not run on hardware
//
#include <hip/hip_runtime.h>

typedef __attribute__((ext_vector_type(16))) _Float16 v16h;
typedef __attribute__((ext_vector_type(8)))  float    v8f;

#define HW 65536   // 256*256 pixels
#define NW 8       // waves per block (256 threads)

union Frag { v16h h; unsigned int u[8]; };

static __device__ __forceinline__ unsigned short f2h(float x) {
    _Float16 h = (_Float16)x;
    return __builtin_bit_cast(unsigned short, h);
}
static __device__ __forceinline__ unsigned int pack2(float a, float b) {
    return (unsigned int)f2h(a) | ((unsigned int)f2h(b) << 16);
}

// ds_swizzle_b32, group-of-32 mode: lane i reads lane (i & and) | or ^ xor.
// XOR-butterfly patterns: offset = (xor_mask << 10) | 0x1f
template <int PAT>
static __device__ __forceinline__ float swz(float x) {
    return __int_as_float(__builtin_amdgcn_ds_swizzle(__float_as_int(x), PAT));
}

__global__ __launch_bounds__(256) void aligner_kernel(
    const float* __restrict__ ego_cls, const float* __restrict__ ego_reg,
    const float* __restrict__ other_cls, const float* __restrict__ other_reg,
    const float* __restrict__ W1, const float* __restrict__ b1,
    const float* __restrict__ gamma, const float* __restrict__ beta,
    const float* __restrict__ W2, const float* __restrict__ b2,
    float* __restrict__ out_cls, float* __restrict__ out_reg)
{
    // f16 weights, col-major [n][k] so B-fragment K-pairs are contiguous dwords
    __shared__ __align__(16) unsigned short sW1[64*32];  // K padded 6->32
    __shared__ __align__(16) unsigned short sW2[64*64];
    __shared__ float sB1[64], sG[64], sBe[64], sB2[64];
    __shared__ __align__(16) unsigned short sHn [NW][16*64]; // LN output, row-major [m][k]
    __shared__ __align__(16) unsigned short sEgo[NW][16*64]; // encoded descriptors
    __shared__ __align__(16) unsigned short sOth[NW][16*64];
    __shared__ float sAttn[NW][4][4][4];                      // [pix][i][j]

    const int tid = threadIdx.x;

    // ---- prefetch this block's HBM streams so L2/WGP$ fill overlaps the GEMMs ----
    {
        int pb = blockIdx.x * (NW*4);               // 32-pixel window (128 B/row)
        if (tid < 64) {                             // all 16 "other" channels x 4 batches
            int j = tid >> 4, c = tid & 15;
            const float* p = (c < 2) ? other_cls + (j*2 + c)*HW + pb
                                     : other_reg + (j*14 + (c-2))*HW + pb;
            __builtin_prefetch(p, 0, 3);
            __builtin_prefetch(p + 16, 0, 3);
        } else if (tid < 88) {                      // ego descriptor channel rows
            int idx = tid - 64;                     // 0..23 = 4 batches x 6 channels
            int j = idx / 6, e = idx % 6;
            const float* p = (e < 4) ? ego_reg + (j*14 + 3 + e)*HW + pb
                                     : ego_cls + (j*2 + (e-4))*HW + pb;
            __builtin_prefetch(p, 0, 3);
            __builtin_prefetch(p + 16, 0, 3);
        }
    }

    // ---- stage weights into LDS (f16) ----
    #pragma unroll
    for (int it = 0; it < 8; ++it) {            // 2048 entries of sW1
        int idx = tid + it*256;
        int n = idx >> 5, k = idx & 31;         // sW1[n*32+k]
        sW1[idx] = (k < 6) ? f2h(W1[k*64 + n]) : (unsigned short)0;
    }
    #pragma unroll
    for (int it = 0; it < 16; ++it) {           // 4096 entries of sW2
        int idx = tid + it*256;
        int n = idx >> 6, k = idx & 63;         // sW2[n*64+k]
        sW2[idx] = f2h(W2[k*64 + n]);
    }
    if (tid < 64) { sB1[tid]=b1[tid]; sG[tid]=gamma[tid]; sBe[tid]=beta[tid]; sB2[tid]=b2[tid]; }
    __syncthreads();

    const int lane = tid & 31;
    const int w    = tid >> 5;
    const int half = lane >> 4;   // which 16-lane group
    const int lrow = lane & 15;
    const int p0   = blockIdx.x * (NW*4) + w*4;   // base pixel of this wave (4 pixels)

    float b1v[4], gv[4], bev[4], b2v[4];
    #pragma unroll
    for (int t = 0; t < 4; ++t) {
        int n = 16*t + lrow;
        b1v[t]=sB1[n]; gv[t]=sG[n]; bev[t]=sBe[n]; b2v[t]=sB2[n];
    }
    const v8f vzero = {0.f,0.f,0.f,0.f,0.f,0.f,0.f,0.f};

    // ================= encode ego (stream 0) and other (stream 1) =================
    for (int stream = 0; stream < 2; ++stream) {
        const float* cls = stream ? other_cls : ego_cls;
        const float* reg = stream ? other_reg : ego_reg;
        unsigned short* dmat = stream ? sOth[w] : sEgo[w];

        // --- descriptor -> A-fragment of GEMM1, built directly in registers ---
        // row m = lane&15 = 4*localpix + batch ; K = 6 real features, padded to 32
        Frag a1;
        #pragma unroll
        for (int v = 0; v < 8; ++v) a1.u[v] = 0u;
        {
            int pl = lrow >> 2, j = lrow & 3;
            int ng = p0 + pl;
            const float* rp = reg + (j*14)*HW + ng;
            const float* cp = cls + (j*2 )*HW + ng;
            float d0 = rp[3*HW], d1 = rp[4*HW], d2 = rp[5*HW];
            float th = rp[6*HW];
            float d3 = __sinf(th), d4 = __cosf(th);
            float d5 = fmaxf(cp[0], cp[HW]);
            if (half == 0) {                    // lanes 0-15 hold K=0..7 in VGPR0..3
                a1.u[0] = pack2(d0, d1);
                a1.u[1] = pack2(d2, d3);
                a1.u[2] = pack2(d4, d5);
            }                                   // K>=6 (and all of lanes 16-31) are zero
        }

        // --- GEMM1: h = desc(16x32) @ W1(32x64), one k-step, 4 N-tiles ---
        v8f h[4];
        #pragma unroll
        for (int t = 0; t < 4; ++t) {
            Frag b;
            #pragma unroll
            for (int v = 0; v < 8; ++v) {
                int ncol = 16*t + lrow;
                int k    = 16*half + 2*v;       // B layout: lanes0-15 K=0..15, lanes16-31 K=16..31
                b.u[v] = *(const unsigned int*)&sW1[ncol*32 + k];
            }
            h[t] = __builtin_amdgcn_wmma_f32_16x16x32_f16(false, a1.h, false, b.h,
                                                          (short)0, vzero, false, false);
        }
        // bias + relu  (C/D layout: row m = r + 8*half, col n = 16*t + lrow)
        #pragma unroll
        for (int t = 0; t < 4; ++t)
            #pragma unroll
            for (int r = 0; r < 8; ++r)
                h[t][r] = fmaxf(h[t][r] + b1v[t], 0.0f);

        // --- LayerNorm over the 64 features of each row ---
        // Row sums live across the 16-lane half: 4-step XOR butterfly with
        // ds_swizzle_b32 (no address VGPRs, immediate pattern).
        #pragma unroll
        for (int r = 0; r < 8; ++r) {
            float s = 0.f, q = 0.f;
            #pragma unroll
            for (int t = 0; t < 4; ++t) { float x = h[t][r]; s += x; q += x*x; }
            s += swz<0x041f>(s);  q += swz<0x041f>(q);   // xor 1
            s += swz<0x081f>(s);  q += swz<0x081f>(q);   // xor 2
            s += swz<0x101f>(s);  q += swz<0x101f>(q);   // xor 4
            s += swz<0x201f>(s);  q += swz<0x201f>(q);   // xor 8
            float mu  = s * (1.0f/64.0f);
            float var = q * (1.0f/64.0f) - mu*mu;
            float rs  = rsqrtf(var + 1e-5f);
            int mrow = r + 8*half;
            #pragma unroll
            for (int t = 0; t < 4; ++t) {
                float x = (h[t][r] - mu) * rs * gv[t] + bev[t];
                sHn[w][mrow*64 + 16*t + lrow] = f2h(x);
            }
        }
        __syncthreads();

        // --- GEMM2: d = LN(h)(16x64) @ W2(64x64) + b2 ; 2 k-steps, 4 N-tiles ---
        v8f d[4];
        #pragma unroll
        for (int t = 0; t < 4; ++t) d[t] = vzero;
        #pragma unroll
        for (int s = 0; s < 2; ++s) {
            Frag a;
            #pragma unroll
            for (int v = 0; v < 8; ++v) {       // A layout: lanes0-15 K=0..15 ; lanes16-31 K=16..31
                int k = 16*half + 2*v + 32*s;
                a.u[v] = *(const unsigned int*)&sHn[w][lrow*64 + k];
            }
            #pragma unroll
            for (int t = 0; t < 4; ++t) {
                Frag b;
                #pragma unroll
                for (int v = 0; v < 8; ++v) {
                    int ncol = 16*t + lrow;
                    int k    = 16*half + 2*v + 32*s;
                    b.u[v] = *(const unsigned int*)&sW2[ncol*64 + k];
                }
                d[t] = __builtin_amdgcn_wmma_f32_16x16x32_f16(false, a.h, false, b.h,
                                                              (short)0, d[t], false, false);
            }
        }
        #pragma unroll
        for (int t = 0; t < 4; ++t)
            #pragma unroll
            for (int r = 0; r < 8; ++r) {
                float x = d[t][r] + b2v[t];
                dmat[(r + 8*half)*64 + 16*t + lrow] = f2h(x);
            }
        __syncthreads();
    }

    // ================= attn(16x16) = ego_d @ oth_d^T / 8 (diag 4x4 blocks used) =================
    v8f at = vzero;
    #pragma unroll
    for (int s = 0; s < 2; ++s) {
        Frag a, b;
        #pragma unroll
        for (int v = 0; v < 8; ++v) {
            int ka = 16*half + 2*v + 32*s;
            a.u[v] = *(const unsigned int*)&sEgo[w][lrow*64 + ka];
            int kb = 16*half + 2*v + 32*s;                 // B[k][n] = oth_d[n][k]
            b.u[v] = *(const unsigned int*)&sOth[w][lrow*64 + kb];
        }
        at = __builtin_amdgcn_wmma_f32_16x16x32_f16(false, a.h, false, b.h,
                                                    (short)0, at, false, false);
    }
    #pragma unroll
    for (int r = 0; r < 8; ++r) {
        int m = r + 8*half, n = lrow;
        if ((m >> 2) == (n >> 2))                          // same pixel -> keep
            sAttn[w][m >> 2][m & 3][n & 3] = at[r] * 0.125f;
    }
    __syncthreads();

    // ================= apply attn to the 16 "other" channels =================
    {
        int pix = lane & 3, i = (lane >> 2) & 3;
        int ng = p0 + pix;
        float a0 = sAttn[w][pix][i][0];
        float a1 = sAttn[w][pix][i][1];
        float a2 = sAttn[w][pix][i][2];
        float a3 = sAttn[w][pix][i][3];
        #pragma unroll
        for (int cc = 0; cc < 8; ++cc) {
            int c = 8*half + cc;                           // 0..15 total channels
            if (c < 2) {
                float acc = a0*other_cls[(0*2+c)*HW+ng] + a1*other_cls[(1*2+c)*HW+ng]
                          + a2*other_cls[(2*2+c)*HW+ng] + a3*other_cls[(3*2+c)*HW+ng];
                out_cls[(i*2 + c)*HW + ng] = acc;
            } else {
                int cr = c - 2;                            // 0..13
                float acc = a0*other_reg[(0*14+cr)*HW+ng] + a1*other_reg[(1*14+cr)*HW+ng]
                          + a2*other_reg[(2*14+cr)*HW+ng] + a3*other_reg[(3*14+cr)*HW+ng];
                out_reg[(i*14 + cr)*HW + ng] = acc;
            }
        }
    }
}

extern "C" void kernel_launch(void* const* d_in, const int* in_sizes, int n_in,
                              void* d_out, int out_size, void* d_ws, size_t ws_size,
                              hipStream_t stream) {
    const float* ego_cls   = (const float*)d_in[0];
    const float* ego_reg   = (const float*)d_in[1];
    const float* other_cls = (const float*)d_in[2];
    const float* other_reg = (const float*)d_in[3];
    const float* W1    = (const float*)d_in[4];
    const float* b1    = (const float*)d_in[5];
    const float* gamma = (const float*)d_in[6];
    const float* beta  = (const float*)d_in[7];
    const float* W2    = (const float*)d_in[8];
    const float* b2    = (const float*)d_in[9];

    float* out     = (float*)d_out;
    float* out_cls = out;                 // (4,2,256,256)
    float* out_reg = out + 4*2*HW;        // (4,14,256,256)

    dim3 grid(HW / (NW*4));               // 2048 blocks, 32 pixels each
    aligner_kernel<<<grid, 256, 0, stream>>>(ego_cls, ego_reg, other_cls, other_reg,
                                             W1, b1, gamma, beta, W2, b2,
                                             out_cls, out_reg);
}
